// VAEDummy2_77352361001091
// MI455X (gfx1250) — compile-verified, matrix operand order checked by reading, not run on hardware
//
#include <hip/hip_runtime.h>
#include <hip/hip_bf16.h>

// ---------------------------------------------------------------------------
// GRU seq2seq forward on gfx1250 (MI455X): bf16 WMMA everywhere, fused GRU
// epilogue, ping-pong h buffers, logits written directly to d_out.
// Staging: double-buffered async global->LDS (ASYNCcnt, in-order completion)
// overlapping the next chunk's DMA with the current chunk's WMMAs. Every wave
// issues a fixed number of async copies per chunk (3 GRU, 2 bias GEMM) -> one uniform s_wait_asynccnt.
// ---------------------------------------------------------------------------

typedef __bf16 bf16_t;
typedef __bf16 bf16x8  __attribute__((ext_vector_type(8)));
typedef __bf16 bf16x16 __attribute__((ext_vector_type(16)));
typedef float  f32x8   __attribute__((ext_vector_type(8)));
typedef int    v4i     __attribute__((vector_size(16)));

static constexpr int kB = 1024;   // batch
static constexpr int kH = 256;    // hidden
static constexpr int kE = 512;    // embedding dim
static constexpr int kS = 64;     // encoder seq len
static constexpr int kV = 5000;   // vocab
static constexpr int kT = 24;     // decoder steps
static constexpr int kG = 768;    // 3*H gate rows
static constexpr int LSTR = 40;   // LDS row stride in halves (32 data + 8 pad)

#define AS1 __attribute__((address_space(1)))
#define AS3 __attribute__((address_space(3)))

#if defined(__has_builtin)
#if __has_builtin(__builtin_amdgcn_global_load_async_to_lds_b128)
#define HAVE_ASYNC_LDS 1
#endif
#endif
#ifndef HAVE_ASYNC_LDS
#define HAVE_ASYNC_LDS 0
#endif

// 16-byte global -> LDS copy. Async path writes LDS directly (no VGPRs, no
// loadcnt stall); fallback is a register-staged b128 copy.
__device__ __forceinline__ void copy16_g2l(const bf16_t* g, bf16_t* l) {
#if HAVE_ASYNC_LDS
  __builtin_amdgcn_global_load_async_to_lds_b128(
      (AS1 v4i*)(uintptr_t)g, (AS3 v4i*)(unsigned)(uintptr_t)l, 0, 0);
#else
  *(bf16x8*)l = *(const bf16x8*)g;
#endif
}

// Wait until this wave's outstanding async copies <= N. Async loads complete
// in order, so waiting <= (count just issued) drains all older chunks.
template <int N>
__device__ __forceinline__ void await_async() {
#if HAVE_ASYNC_LDS
#if __has_builtin(__builtin_amdgcn_s_wait_asynccnt)
  __builtin_amdgcn_s_wait_asynccnt(N);
#else
  asm volatile("s_wait_asynccnt %0" ::"i"(N) : "memory");
#endif
#endif
}

__device__ __forceinline__ float sigmoidf_(float x) {
  return 1.0f / (1.0f + __expf(-x));
}

// Load one wave-lane's WMMA fragment from an LDS tile laid out row-major with
// stride LSTR halves. Per ISA (CDNA5 ISA, 16-bit A 16x32 layout):
//   lane half = lane>>4 selects K {0..7,16..23} (half 0) or {8..15,24..31}.
// The B fragment (32x16, K-major source W[n,k]) has the symmetric layout.
__device__ __forceinline__ bf16x16 load_frag(const bf16_t* tile, int row, int half) {
  union { bf16x16 v; bf16x8 h[2]; } u;
  const bf16_t* p = tile + row * LSTR;
  u.h[0] = *(const bf16x8*)(p + half * 8);
  u.h[1] = *(const bf16x8*)(p + 16 + half * 8);
  return u.v;
}

__device__ __forceinline__ f32x8 wmma_bf16(bf16x16 a, bf16x16 b, f32x8 c) {
  return __builtin_amdgcn_wmma_f32_16x16x32_bf16(
      /*neg_a=*/false, a, /*neg_b=*/false, b,
      /*c_mod=*/(short)0, c, /*reuse_a=*/false, /*reuse_b=*/false);
}

// ---------------------------------------------------------------------------
// Elementwise / prep kernels
// ---------------------------------------------------------------------------

__global__ void cast_bf16_kernel(const float* __restrict__ src,
                                 bf16_t* __restrict__ dst, int n) {
  int i = blockIdx.x * blockDim.x + threadIdx.x;
  if (i < n) dst[i] = (bf16_t)src[i];
}

// wcat[row, 0..K1) = w_ih[row, :], wcat[row, K1..K1+K2) = w_hh[row, :]
__global__ void build_wcat_kernel(const float* __restrict__ wih,
                                  const float* __restrict__ whh,
                                  bf16_t* __restrict__ wcat, int K1, int K2) {
  int K = K1 + K2;
  int i = blockIdx.x * blockDim.x + threadIdx.x;
  if (i >= kG * K) return;
  int row = i / K, col = i - row * K;
  float v = (col < K1) ? wih[row * K1 + col] : whh[row * K2 + (col - K1)];
  wcat[i] = (bf16_t)v;
}

__global__ void zero_bf16_rows_kernel(bf16_t* __restrict__ dst, int ld, int width) {
  int segs = width >> 3;
  int i = blockIdx.x * blockDim.x + threadIdx.x;
  if (i >= kB * segs) return;
  int b = i / segs, seg = i - b * segs;
  bf16x8 z;
#pragma unroll
  for (int j = 0; j < 8; ++j) z[j] = (bf16_t)0.0f;
  *(bf16x8*)&dst[(long)b * ld + seg * 8] = z;
}

__global__ void zero_int_kernel(int* __restrict__ p, int n) {
  int i = blockIdx.x * blockDim.x + threadIdx.x;
  if (i < n) p[i] = 0;
}

// dst[b, 0..E) = emb_bf16[ids[b*id_stride], :]  (E = 512, 16B per thread)
__global__ void gather_emb_kernel(const int* __restrict__ ids, int id_stride,
                                  const bf16_t* __restrict__ emb,
                                  bf16_t* __restrict__ dst, int ld_dst) {
  int i = blockIdx.x * blockDim.x + threadIdx.x;   // kB * (kE/8) threads
  int b = i >> 6, seg = i & 63;
  int id = ids[b * id_stride];
  *(bf16x8*)&dst[(long)b * ld_dst + seg * 8] =
      *(const bf16x8*)&emb[(long)id * kE + seg * 8];
}

// Row argmax over N f32 values (first-max tie break like jnp.argmax).
__global__ __launch_bounds__(256)
void argmax_kernel(const float* __restrict__ logits, long row_stride, int N,
                   int* __restrict__ ids) {
  __shared__ float sv[256];
  __shared__ int   si[256];
  int b = blockIdx.x;
  const float* row = logits + (long)b * row_stride;
  float best = -3.402823466e+38f;
  int   bi   = 0x7fffffff;
  for (int i = threadIdx.x; i < N; i += 256) {
    float v = row[i];
    if (v > best || (v == best && i < bi)) { best = v; bi = i; }
  }
  sv[threadIdx.x] = best; si[threadIdx.x] = bi;
  __syncthreads();
  for (int s = 128; s > 0; s >>= 1) {
    if (threadIdx.x < s) {
      float ov = sv[threadIdx.x + s]; int oi = si[threadIdx.x + s];
      if (ov > sv[threadIdx.x] || (ov == sv[threadIdx.x] && oi < si[threadIdx.x])) {
        sv[threadIdx.x] = ov; si[threadIdx.x] = oi;
      }
    }
    __syncthreads();
  }
  if (threadIdx.x == 0) ids[b] = si[0];
}

// ---------------------------------------------------------------------------
// Fused GEMM + GRU cell.
//   A = [x_t | h_{t-1}]  (kB x K bf16, lda), K1 = split point (x width)
//   wcat = [W_ih | W_hh] (kG x K bf16)
//   r/z gates accumulate over all K; the n-gate splits at K1 into i_n and h_n
//   accumulators (two sequential K loops -> no per-iteration branch).
// 8 waves per WG as 4(M) x 2(N): WG tile = 64 rows x 32 h-cols (x3 gates).
// Double-buffered LDS: chunk c+1's async DMA overlaps chunk c's WMMAs.
// Grid: (kB/64, kH/32) = (16, 8).
// ---------------------------------------------------------------------------
__global__ __launch_bounds__(256)
void gemm_gru_kernel(const bf16_t* __restrict__ A, int lda,
                     const bf16_t* __restrict__ wcat,
                     const float* __restrict__ b_ih,
                     const float* __restrict__ b_hh,
                     int K, int K1,
                     bf16_t* __restrict__ h_self, int ld_self,
                     bf16_t* __restrict__ h_next, int ld_next) {
  __shared__ bf16_t lA[2][64 * LSTR];
  __shared__ bf16_t lB[2][96 * LSTR];   // 3 gates x 32 rows per buffer

  const int tid  = threadIdx.x;
  const int wave = tid >> 5;
  const int lane = tid & 31;
  const int wm = wave >> 1, wn = wave & 1;
  const int mbase = blockIdx.x * 64;
  const int hbase = blockIdx.y * 32;
  const int half = lane >> 4, idx = lane & 15;

  // Fixed staging, 16B per item, 3 async instructions per wave per chunk:
  //   A : item tid                (rows 0..63)
  //   B0: item tid                (lB rows 0..63, gates r/z)
  //   B1: 128 items j = wave*16 + lane, lanes 0..15 of every wave
  //       (lB rows 64..95, gate n) -> half-wave EXEC, still 1 instr/wave.
  const int srow = tid >> 2, sseg = tid & 3;
  const int j1   = wave * 16 + idx;               // 0..127
  const int b1row = 64 + (j1 >> 2), b1seg = j1 & 3;
  const bool b1ok = (half == 0);                  // lanes 0..15
  const bf16_t* gA  = A + (long)(mbase + srow) * lda + sseg * 8;
  const bf16_t* gB0 = wcat + (long)((srow >> 5) * kH + hbase + (srow & 31)) * K + sseg * 8;
  const bf16_t* gB1 = wcat + (long)(2 * kH + hbase + ((b1row - 64) & 31)) * K + b1seg * 8;
  const int aoff  = srow * LSTR + sseg * 8;
  const int b1off = b1row * LSTR + b1seg * 8;

  auto issue = [&](int kc, int buf) {
    copy16_g2l(gA + kc, &lA[buf][aoff]);
    copy16_g2l(gB0 + kc, &lB[buf][aoff]);
    if (b1ok) copy16_g2l(gB1 + kc, &lB[buf][b1off]);
  };

  f32x8 acc_r, acc_z, acc_in, acc_hn;
#pragma unroll
  for (int i = 0; i < 8; ++i) { acc_r[i] = 0.f; acc_z[i] = 0.f; acc_in[i] = 0.f; acc_hn[i] = 0.f; }

  const int nc = K >> 5, c1 = K1 >> 5;
  int c = 0;
  issue(0, 0);

  auto step = [&](f32x8& acc_n) {
    const int buf = c & 1;
    if (c + 1 < nc) {
      issue((c + 1) << 5, buf ^ 1);
      if (c + 2 < nc) __builtin_prefetch(gA + ((c + 2) << 5), 0, 1);
      await_async<3>();                            // older chunk has landed
    } else {
      await_async<0>();
    }
    __syncthreads();                               // chunk c landed (all waves)

    bf16x16 a  = load_frag(lA[buf], wm * 16 + idx, half);
    bf16x16 br = load_frag(lB[buf],      wn * 16 + idx, half);
    bf16x16 bz = load_frag(lB[buf], 32 + wn * 16 + idx, half);
    bf16x16 bn = load_frag(lB[buf], 64 + wn * 16 + idx, half);
    acc_r = wmma_bf16(a, br, acc_r);
    acc_z = wmma_bf16(a, bz, acc_z);
    acc_n = wmma_bf16(a, bn, acc_n);
    __syncthreads();                               // buf free for reuse
    ++c;
  };
  while (c < c1) step(acc_in);
  while (c < nc) step(acc_hn);

  // Fused GRU epilogue. Acc layout: elem i, lane -> m = i + half*8, n = idx.
  const int n  = hbase + wn * 16 + idx;
  const float br_ = b_ih[n]       + b_hh[n];
  const float bz_ = b_ih[kH + n]  + b_hh[kH + n];
  const float bin = b_ih[2 * kH + n];
  const float bhn = b_hh[2 * kH + n];
#pragma unroll
  for (int i = 0; i < 8; ++i) {
    int m = mbase + wm * 16 + i + half * 8;
    float r  = sigmoidf_(acc_r[i] + br_);
    float zg = sigmoidf_(acc_z[i] + bz_);
    float nn = tanhf(acc_in[i] + bin + r * (acc_hn[i] + bhn));
    float hp = (float)A[(long)m * lda + K1 + n];   // h_{t-1} lives in A's tail
    float hv = (1.0f - zg) * nn + zg * hp;
    bf16_t hb = (bf16_t)hv;
    h_self[(long)m * ld_self + n] = hb;
    if (h_next) h_next[(long)m * ld_next + n] = hb;
  }
}

// ---------------------------------------------------------------------------
// Generic bf16 GEMM: C = A(MxK) @ W(NxK)^T + bias, f32 and/or bf16 outputs.
// 8 waves as 4(M) x 2(N-pairs): WG tile 64x64, each wave 16x32 (2 accums).
// OOB W rows are clamped to row N-1 (their products only land in accumulator
// columns that the guarded store never writes). Double-buffered async LDS.
// Grid: (M/64, ceil(N/64)).
// ---------------------------------------------------------------------------
__global__ __launch_bounds__(256)
void gemm_bias_kernel(const bf16_t* __restrict__ A, int lda,
                      const bf16_t* __restrict__ W, int ldw,
                      const float* __restrict__ bias,
                      float* __restrict__ Cf, long ldc,
                      bf16_t* __restrict__ Cb, int ldcb,
                      int N, int K) {
  __shared__ bf16_t lA[2][64 * LSTR];
  __shared__ bf16_t lB[2][64 * LSTR];

  const int tid  = threadIdx.x;
  const int wave = tid >> 5;
  const int lane = tid & 31;
  const int wm = wave >> 1, wn = wave & 1;
  const int mbase = blockIdx.x * 64;
  const int nbase = blockIdx.y * 64;
  const int half = lane >> 4, idx = lane & 15;

  const int srow = tid >> 2, sseg = tid & 3;
  const int wrow = min(nbase + srow, N - 1);       // clamp OOB rows
  const bf16_t* gA = A + (long)(mbase + srow) * lda + sseg * 8;
  const bf16_t* gW = W + (long)wrow * ldw + sseg * 8;
  const int soff = srow * LSTR + sseg * 8;

  auto issue = [&](int kc, int buf) {
    copy16_g2l(gA + kc, &lA[buf][soff]);
    copy16_g2l(gW + kc, &lB[buf][soff]);
  };

  f32x8 acc0, acc1;
#pragma unroll
  for (int i = 0; i < 8; ++i) { acc0[i] = 0.f; acc1[i] = 0.f; }

  const int nc = K >> 5;
  issue(0, 0);
  for (int c = 0; c < nc; ++c) {
    const int buf = c & 1;
    if (c + 1 < nc) {
      issue((c + 1) << 5, buf ^ 1);
      if (c + 2 < nc) __builtin_prefetch(gW + ((c + 2) << 5), 0, 1);
      await_async<2>();
    } else {
      await_async<0>();
    }
    __syncthreads();

    bf16x16 a  = load_frag(lA[buf], wm * 16 + idx, half);
    bf16x16 b0 = load_frag(lB[buf], wn * 32 + idx, half);
    bf16x16 b1 = load_frag(lB[buf], wn * 32 + 16 + idx, half);
    acc0 = wmma_bf16(a, b0, acc0);
    acc1 = wmma_bf16(a, b1, acc1);
    __syncthreads();
  }

#pragma unroll
  for (int j = 0; j < 2; ++j) {
    const f32x8& acc = j ? acc1 : acc0;
    int nn = nbase + wn * 32 + j * 16 + idx;
    if (nn >= N) continue;
    float bv = bias ? bias[nn] : 0.0f;
#pragma unroll
    for (int i = 0; i < 8; ++i) {
      int m = mbase + wm * 16 + i + half * 8;
      float v = acc[i] + bv;
      if (Cf) Cf[(long)m * ldc + nn] = v;
      if (Cb) Cb[(long)m * ldcb + nn] = (bf16_t)v;
    }
  }
}

// ---------------------------------------------------------------------------
// Host orchestration
// ---------------------------------------------------------------------------
extern "C" void kernel_launch(void* const* d_in, const int* in_sizes, int n_in,
                              void* d_out, int out_size, void* d_ws, size_t ws_size,
                              hipStream_t stream) {
  const float* emb_f    = (const float*)d_in[0];
  const float* e_wih0   = (const float*)d_in[1];
  const float* e_whh0   = (const float*)d_in[2];
  const float* e_bih0   = (const float*)d_in[3];
  const float* e_bhh0   = (const float*)d_in[4];
  const float* e_wih1   = (const float*)d_in[5];
  const float* e_whh1   = (const float*)d_in[6];
  const float* e_bih1   = (const float*)d_in[7];
  const float* e_bhh1   = (const float*)d_in[8];
  const float* tol_w    = (const float*)d_in[9];
  const float* tol_b    = (const float*)d_in[10];
  const float* l2h_w    = (const float*)d_in[11];
  const float* l2h_b    = (const float*)d_in[12];
  const float* d_wih0   = (const float*)d_in[13];
  const float* d_whh0   = (const float*)d_in[14];
  const float* d_bih0   = (const float*)d_in[15];
  const float* d_bhh0   = (const float*)d_in[16];
  const float* d_wih1   = (const float*)d_in[17];
  const float* d_whh1   = (const float*)d_in[18];
  const float* d_bih1   = (const float*)d_in[19];
  const float* d_bhh1   = (const float*)d_in[20];
  const float* fco_w    = (const float*)d_in[21];
  const float* fco_b    = (const float*)d_in[22];
  const int*   x        = (const int*)d_in[23];
  float* out = (float*)d_out;
  float* out_logits = out;                               // [kB, kT, kV]
  float* out_z      = out + (size_t)kB * kT * kV;        // [kB, kH]

  // Workspace carve-up (256B aligned).
  size_t o = 0;
  auto carve = [&](size_t bytes) {
    size_t r = o; o += (bytes + 255) & ~(size_t)255; return r;
  };
  char* ws = (char*)d_ws;
  bf16_t* emb_bf   = (bf16_t*)(ws + carve((size_t)kV * kE * 2));
  bf16_t* wcat_e0  = (bf16_t*)(ws + carve((size_t)kG * (kE + kH) * 2));
  bf16_t* wcat_e1  = (bf16_t*)(ws + carve((size_t)kG * (kH + kH) * 2));
  bf16_t* wcat_d0  = (bf16_t*)(ws + carve((size_t)kG * (kE + kH) * 2));
  bf16_t* wcat_d1  = (bf16_t*)(ws + carve((size_t)kG * (kH + kH) * 2));
  bf16_t* tol_bf   = (bf16_t*)(ws + carve((size_t)kH * kH * 2));
  bf16_t* l2h_bf   = (bf16_t*)(ws + carve((size_t)2 * kH * kH * 2));
  bf16_t* fco_bf   = (bf16_t*)(ws + carve((size_t)kV * kH * 2));
  bf16_t* acat_e0[2], *acat_e1[2], *acat_d0[2], *acat_d1[2];
  for (int i = 0; i < 2; ++i) acat_e0[i] = (bf16_t*)(ws + carve((size_t)kB * (kE + kH) * 2));
  for (int i = 0; i < 2; ++i) acat_e1[i] = (bf16_t*)(ws + carve((size_t)kB * (kH + kH) * 2));
  for (int i = 0; i < 2; ++i) acat_d0[i] = (bf16_t*)(ws + carve((size_t)kB * (kE + kH) * 2));
  for (int i = 0; i < 2; ++i) acat_d1[i] = (bf16_t*)(ws + carve((size_t)kB * (kH + kH) * 2));
  bf16_t* z_bf = (bf16_t*)(ws + carve((size_t)kB * kH * 2));
  int*    ids  = (int*)(ws + carve((size_t)kB * 4));

  auto launch_cast = [&](const float* s, bf16_t* d, int n) {
    cast_bf16_kernel<<<(n + 255) / 256, 256, 0, stream>>>(s, d, n);
  };

  // ---- one-time weight prep ----
  launch_cast(emb_f, emb_bf, kV * kE);
  launch_cast(tol_w, tol_bf, kH * kH);
  launch_cast(l2h_w, l2h_bf, 2 * kH * kH);
  launch_cast(fco_w, fco_bf, kV * kH);
  {
    int n0 = kG * (kE + kH), n1 = kG * (kH + kH);
    build_wcat_kernel<<<(n0 + 255) / 256, 256, 0, stream>>>(e_wih0, e_whh0, wcat_e0, kE, kH);
    build_wcat_kernel<<<(n1 + 255) / 256, 256, 0, stream>>>(e_wih1, e_whh1, wcat_e1, kH, kH);
    build_wcat_kernel<<<(n0 + 255) / 256, 256, 0, stream>>>(d_wih0, d_whh0, wcat_d0, kE, kH);
    build_wcat_kernel<<<(n1 + 255) / 256, 256, 0, stream>>>(d_wih1, d_whh1, wcat_d1, kH, kH);
  }

  // ---- encoder init: zero h tails of ping buffer 0 ----
  {
    int n = kB * (kH / 8);
    zero_bf16_rows_kernel<<<(n + 255) / 256, 256, 0, stream>>>(acat_e0[0] + kE, kE + kH, kH);
    zero_bf16_rows_kernel<<<(n + 255) / 256, 256, 0, stream>>>(acat_e1[0] + kH, kH + kH, kH);
  }

  const dim3 gru_grid(kB / 64, kH / 32);
  const int  gat_thr = kB * (kE / 8);

  // ---- encoder: 64 steps, two stacked GRU layers ----
  for (int t = 0; t < kS; ++t) {
    int pt = t & 1, po = (t + 1) & 1;
    gather_emb_kernel<<<gat_thr / 256, 256, 0, stream>>>(
        x + t, kS, emb_bf, acat_e0[pt], kE + kH);
    gemm_gru_kernel<<<gru_grid, 256, 0, stream>>>(
        acat_e0[pt], kE + kH, wcat_e0, e_bih0, e_bhh0, kE + kH, kE,
        acat_e0[po] + kE, kE + kH,          // self (next step's h tail)
        acat_e1[pt], kH + kH);              // layer-1 input head (this step)
    gemm_gru_kernel<<<gru_grid, 256, 0, stream>>>(
        acat_e1[pt], kH + kH, wcat_e1, e_bih1, e_bhh1, kH + kH, kH,
        acat_e1[po] + kH, kH + kH,
        nullptr, 0);
  }
  // hT of layer 1 now in acat_e1[(kS)&1 = 0] tail.
  bf16_t* hT = acat_e1[0] + kH;

  // ---- latent: z = hT @ tol^T + b  (f32 -> d_out tail, bf16 -> z_bf) ----
  gemm_bias_kernel<<<dim3(kB / 64, kH / 64), 256, 0, stream>>>(
      hT, kH + kH, tol_bf, kH, tol_b, out_z, kH, z_bf, kH, kH, kH);
  // h0 = z @ l2h[0:H]^T + b[0:H]   -> decoder L0 h tail (ping 0)
  gemm_bias_kernel<<<dim3(kB / 64, kH / 64), 256, 0, stream>>>(
      z_bf, kH, l2h_bf, kH, l2h_b, nullptr, 0,
      acat_d0[0] + kE, kE + kH, kH, kH);
  // h1 = z @ l2h[H:2H]^T + b[H:2H] -> decoder L1 h tail (ping 0)
  gemm_bias_kernel<<<dim3(kB / 64, kH / 64), 256, 0, stream>>>(
      z_bf, kH, l2h_bf + (size_t)kH * kH, kH, l2h_b + kH, nullptr, 0,
      acat_d1[0] + kH, kH + kH, kH, kH);

  // ---- decoder: 24 greedy steps ----
  zero_int_kernel<<<(kB + 255) / 256, 256, 0, stream>>>(ids, kB);
  gather_emb_kernel<<<gat_thr / 256, 256, 0, stream>>>(
      ids, 1, emb_bf, acat_d0[0], kE + kH);

  const dim3 fc_grid(kB / 64, (kV + 63) / 64);
  for (int t = 0; t < kT; ++t) {
    int pt = t & 1, po = (t + 1) & 1;
    gemm_gru_kernel<<<gru_grid, 256, 0, stream>>>(
        acat_d0[pt], kE + kH, wcat_d0, d_bih0, d_bhh0, kE + kH, kE,
        acat_d0[po] + kE, kE + kH,
        acat_d1[pt], kH + kH);
    gemm_gru_kernel<<<gru_grid, 256, 0, stream>>>(
        acat_d1[pt], kH + kH, wcat_d1, d_bih1, d_bhh1, kH + kH, kH,
        acat_d1[po] + kH, kH + kH,
        nullptr, 0);
    // logits[:, t, :] = h1 @ fc^T + b  (h1 just written into acat_d1[po] tail)
    gemm_bias_kernel<<<fc_grid, 256, 0, stream>>>(
        acat_d1[po] + kH, kH + kH, fco_bf, kH, fco_b,
        out_logits + (size_t)t * kV, (long)kT * kV, nullptr, 0, kV, kH);
    if (t + 1 < kT) {
      argmax_kernel<<<kB, 256, 0, stream>>>(
          out_logits + (size_t)t * kV, (long)kT * kV, kV, ids);
      gather_emb_kernel<<<gat_thr / 256, 256, 0, stream>>>(
          ids, 1, emb_bf, acat_d0[po], kE + kH);
    }
  }
}
